// MultiHeadAttention_80324478369829
// MI455X (gfx1250) — matrix operand order repaired and checked by reading, not compiled, not run on hardware
//
#include <hip/hip_runtime.h>

typedef __bf16 bf16;
typedef __attribute__((ext_vector_type(16))) __bf16 v16bf;
typedef __attribute__((ext_vector_type(8)))  __bf16 v8bf;
typedef __attribute__((ext_vector_type(8)))  float  v8f;
typedef __attribute__((ext_vector_type(4)))  unsigned int v4u;
typedef __attribute__((ext_vector_type(8)))  int v8i;
typedef __attribute__((ext_vector_type(4)))  int v4i;

#define B_  2
#define S_  2048
#define D_  1024
#define H_  16
#define HD_ 64
#define M_  (B_ * S_)

#if __has_builtin(__builtin_amdgcn_tensor_load_to_lds) && \
    __has_builtin(__builtin_amdgcn_s_wait_tensorcnt)
#define HAVE_TDM 1
#else
#define HAVE_TDM 0
#endif

// Assemble a 16x32 bf16 A-fragment for v_wmma_f32_16x16x32_bf16.
// Contiguous K order (matched by B loads, not ISA table order): lane L (row = L%16) holds K = base+0..15,
// base = (L<16) ? 0 : 16.  Caller passes p already offset by `base`.
__device__ __forceinline__ v16bf load_afrag(const bf16* p) {
  v8bf lo = *(const v8bf*)(p);
  v8bf hi = *(const v8bf*)(p + 8);
  return __builtin_shufflevector(lo, hi, 0, 1, 2, 3, 4, 5, 6, 7,
                                          8, 9, 10, 11, 12, 13, 14, 15);
}

__device__ __forceinline__ v8f wmma_bf16(v16bf a, v16bf b, v8f c) {
  return __builtin_amdgcn_wmma_f32_16x16x32_bf16(
      /*neg_a=*/false, a, /*neg_b=*/false, b,
      /*c_mod=*/(short)0, c, /*reuse_a=*/false, /*reuse_b=*/false);
}

#if HAVE_TDM
// Issue a TDM load of a 64x64 bf16 tile (row stride HD_ elems) from `gsrc`
// into LDS at `lds_byte_off`.  D# layout per CDNA5 ISA ch.8:
//   group0: count=1 | lds_addr | global_addr[56:0] | type=2
//   group1: data_size=2B, tensor=[dim0=64, dim1=2048], tile=[64,64], stride0=64
// This toolchain declares the 6-arg builtin:
//   (uint32x4 g0, int32x8 g1, int32x4, int32x4, int32x8, i32 cpol)
__device__ __forceinline__ void tdm_load_tile_64x64(const bf16* gsrc,
                                                    unsigned lds_byte_off) {
  unsigned long long ga = (unsigned long long)(const void*)gsrc;
  v4u g0;
  g0[0] = 1u;                                            // count=1, user mode
  g0[1] = lds_byte_off;                                  // lds_addr
  g0[2] = (unsigned)ga;                                  // global_addr[31:0]
  g0[3] = (unsigned)((ga >> 32) & 0x1FFFFFFull) | (2u << 30);  // addr | type=2
  v8i g1;
  g1[0] = 0x00010000;          // workgroup_mask=0, data_size=1 (2 bytes)
  g1[1] = (int)(64u << 16);    // [63:48] tensor_dim0 = 64 (HD)
  g1[2] = (int)(2048u << 16);  // [95:80] tensor_dim1 = 2048 (S)
  g1[3] = (int)(64u << 16);    // [127:112] tile_dim0 = 64
  g1[4] = 64;                  // [143:128] tile_dim1 = 64
  g1[5] = 64;                  // [191:160] tensor_dim0_stride = 64
  g1[6] = 0;
  g1[7] = 0;
  v4i z4 = {0, 0, 0, 0};       // 2D tensor: groups 2/3 unused
  v8i z8 = {0, 0, 0, 0, 0, 0, 0, 0};
  __builtin_amdgcn_tensor_load_to_lds(g0, g1, z4, z4, z8, 0);
}
#endif

// ---------------------------------------------------------------------------
// Kernel 1: fused QKV projection.  y = x @ W^T + bias, output stored bf16 in
// [B, H, S, HD] layout.  blockIdx.z selects which of Q/K/V we compute.
// 128 threads = 4 waves; block computes a 64(M) x 64(N) tile; K staged 32 wide.
// ---------------------------------------------------------------------------
__global__ __launch_bounds__(128) void qkv_proj_kernel(
    const float* __restrict__ xq, const float* __restrict__ xk,
    const float* __restrict__ xv,
    const float* __restrict__ wq, const float* __restrict__ bq,
    const float* __restrict__ wk, const float* __restrict__ bk,
    const float* __restrict__ wv, const float* __restrict__ bv,
    bf16* __restrict__ oq, bf16* __restrict__ ok, bf16* __restrict__ ov) {
  __shared__ __align__(32) bf16 lx[64][32];
  __shared__ __align__(32) bf16 lw[64][32];

  const float* X; const float* W; const float* bias; bf16* dst;
  if (blockIdx.z == 0)      { X = xq; W = wq; bias = bq; dst = oq; }
  else if (blockIdx.z == 1) { X = xk; W = wk; bias = bk; dst = ok; }
  else                      { X = xv; W = wv; bias = bv; dst = ov; }

  const int m0   = blockIdx.x * 64;
  const int n0   = blockIdx.y * 64;
  const int tid  = threadIdx.x;
  const int wave = tid >> 5;
  const int lane = tid & 31;

  v8f acc[4] = {};  // four 16x16 column tiles for this wave's 16 rows

  // cooperative staging map: thread t -> row t/2, half-row (t&1)*16
  const int rs = tid >> 1;
  const int ks = (tid & 1) * 16;

  for (int k0 = 0; k0 < D_; k0 += 32) {
    const float* srcx = X + (size_t)(m0 + rs) * D_ + k0 + ks;
    const float* srcw = W + (size_t)(n0 + rs) * D_ + k0 + ks;
#pragma unroll
    for (int i = 0; i < 4; ++i) {
      float4 fx = ((const float4*)srcx)[i];
      float4 fw = ((const float4*)srcw)[i];
      lx[rs][ks + 4 * i + 0] = (bf16)fx.x; lx[rs][ks + 4 * i + 1] = (bf16)fx.y;
      lx[rs][ks + 4 * i + 2] = (bf16)fx.z; lx[rs][ks + 4 * i + 3] = (bf16)fx.w;
      lw[rs][ks + 4 * i + 0] = (bf16)fw.x; lw[rs][ks + 4 * i + 1] = (bf16)fw.y;
      lw[rs][ks + 4 * i + 2] = (bf16)fw.z; lw[rs][ks + 4 * i + 3] = (bf16)fw.w;
    }
    __syncthreads();

    const int rowA  = wave * 16 + (lane & 15);
    const int baseA = (lane < 16) ? 0 : 16;
    v16bf afrag = load_afrag(&lx[rowA][baseA]);

    // B-fragment: lane n holds 16 consecutive K of column n; column n of
    // B = W^T is row n of W -> contiguous in lw.  Load all four fragments
    // first so the WMMAs issue back-to-back behind one s_wait_dscnt.
    const int koffB = (lane < 16) ? 0 : 16;
    v16bf bfrag[4];
#pragma unroll
    for (int t = 0; t < 4; ++t)
      bfrag[t] = *(const v16bf*)&lw[t * 16 + (lane & 15)][koffB];
#pragma unroll
    for (int t = 0; t < 4; ++t)
      acc[t] = wmma_bf16(afrag, bfrag[t], acc[t]);
    __syncthreads();
  }

  // bias + store bf16 to [B, H, S, HD]
#pragma unroll
  for (int t = 0; t < 4; ++t) {
    const int ncol = n0 + t * 16 + (lane & 15);
    const float bval = bias[ncol];
    const int h  = ncol >> 6;
    const int hd = ncol & 63;
#pragma unroll
    for (int rr = 0; rr < 8; ++rr) {
      const int mrow = m0 + wave * 16 + rr + ((lane >> 4) << 3);
      const int bb = mrow >> 11;     // / S_
      const int ss = mrow & 2047;    // % S_
      const float val = acc[t][rr] + bval;
      dst[(((size_t)bb * H_ + h) * S_ + ss) * HD_ + hd] = (bf16)val;
    }
  }
}

// ---------------------------------------------------------------------------
// Kernel 2: flash attention over one (b, h, 64-query block).
// 4 waves x 16 query rows.  K tile via TDM (tensor_load_to_lds) when
// available; V staged transposed so P@V B-fragments are contiguous LDS loads.
// ---------------------------------------------------------------------------
__global__ __launch_bounds__(128) void flash_attn_kernel(
    const bf16* __restrict__ wsq, const bf16* __restrict__ wsk,
    const bf16* __restrict__ wsv,
    float* __restrict__ ctx,    // [B, H, S, HD]
    float* __restrict__ attn) { // [B, S, D]
  __shared__ __align__(32) bf16 lk[64][64];     // K tile, row-major [kv][hd]
  __shared__ __align__(32) bf16 lvT[64][64];    // V tile, transposed [hd][kv]
  __shared__ __align__(32) bf16 lp[4][16][64];  // per-wave P scratch

  const int qb   = blockIdx.x;
  const int h    = blockIdx.y;
  const int b    = blockIdx.z;
  const int tid  = threadIdx.x;
  const int wave = tid >> 5;
  const int lane = tid & 31;

  const size_t headoff = (((size_t)b * H_ + h) * S_) * HD_;
  const bf16* Qh = wsq + headoff;
  const bf16* Kh = wsk + headoff;
  const bf16* Vh = wsv + headoff;

  // Q fragments held in registers for the whole kernel (HD=64 -> 2 frags)
  const int rowA  = qb * 64 + wave * 16 + (lane & 15);
  const int baseA = (lane < 16) ? 0 : 16;
  v16bf qfrag[2];
#pragma unroll
  for (int c = 0; c < 2; ++c)
    qfrag[c] = load_afrag(Qh + (size_t)rowA * HD_ + c * 32 + baseA);

  float m8[8], l8[8];
  v8f oacc[4] = {};
#pragma unroll
  for (int i = 0; i < 8; ++i) { m8[i] = -3.0e38f; l8[i] = 0.0f; }

  // cooperative staging map: thread t -> kv row t/2, hd half (t&1)*32
  const int rcp = tid >> 1;
  const int ccp = (tid & 1) * 32;

  for (int kv0 = 0; kv0 < S_; kv0 += 64) {
    // ---- K tile: TDM DMA into LDS (wave 0 issues; EXEC-independent) ----
#if HAVE_TDM
    if (wave == 0)
      tdm_load_tile_64x64(Kh + (size_t)kv0 * HD_,
                          (unsigned)(size_t)(&lk[0][0]));
#else
    {
      const uint4* sk = (const uint4*)(Kh + (size_t)(kv0 + rcp) * HD_ + ccp);
      uint4* dk = (uint4*)&lk[rcp][ccp];
#pragma unroll
      for (int i = 0; i < 4; ++i) dk[i] = sk[i];
    }
#endif
    // ---- V tile: cooperative transposed staging ----
    {
      const uint4* sv = (const uint4*)(Vh + (size_t)(kv0 + rcp) * HD_ + ccp);
      bf16 tmp[32];
      *(uint4*)&tmp[0]  = sv[0];
      *(uint4*)&tmp[8]  = sv[1];
      *(uint4*)&tmp[16] = sv[2];
      *(uint4*)&tmp[24] = sv[3];
#pragma unroll
      for (int i = 0; i < 32; ++i) lvT[ccp + i][rcp] = tmp[i];
    }
    if (kv0 + 64 < S_) {
      __builtin_prefetch(Vh + (size_t)(kv0 + 64 + rcp) * HD_ + ccp, 0, 1);
#if !HAVE_TDM
      __builtin_prefetch(Kh + (size_t)(kv0 + 64 + rcp) * HD_ + ccp, 0, 1);
#endif
    }
#if HAVE_TDM
    if (wave == 0) __builtin_amdgcn_s_wait_tensorcnt(0);
#endif
    __syncthreads();

    // ---- S = Q @ K^T : load all 8 B-fragments, then 8 WMMAs ----
    v8f sacc[4] = {};
    const int koffB = (lane < 16) ? 0 : 16;
    v16bf kf[8];
#pragma unroll
    for (int t = 0; t < 4; ++t) {
      const int n = t * 16 + (lane & 15);  // kv position = column of B
#pragma unroll
      for (int c = 0; c < 2; ++c)
        kf[t * 2 + c] = *(const v16bf*)&lk[n][c * 32 + koffB];
    }
#pragma unroll
    for (int t = 0; t < 4; ++t)
#pragma unroll
      for (int c = 0; c < 2; ++c)
        sacc[t] = wmma_bf16(qfrag[c], kf[t * 2 + c], sacc[t]);

    // ---- online softmax; vgpr rr <-> row rr + 8*(lane>=16) ----
    const float scale = 0.125f;  // 1/sqrt(HD)
#pragma unroll
    for (int rr = 0; rr < 8; ++rr) {
      float rmax = fmaxf(fmaxf(sacc[0][rr], sacc[1][rr]),
                         fmaxf(sacc[2][rr], sacc[3][rr]));
#pragma unroll
      for (int off = 1; off < 16; off <<= 1)
        rmax = fmaxf(rmax, __shfl_xor(rmax, off, 32));
      rmax *= scale;
      const float mnew = fmaxf(m8[rr], rmax);
      const float corr = __expf(m8[rr] - mnew);
      float psum = 0.0f;
      const int rowIdx = rr + ((lane >> 4) << 3);
#pragma unroll
      for (int t = 0; t < 4; ++t) {
        float p = __expf(sacc[t][rr] * scale - mnew);
        psum += p;
        lp[wave][rowIdx][t * 16 + (lane & 15)] = (bf16)p;
      }
#pragma unroll
      for (int off = 1; off < 16; off <<= 1)
        psum += __shfl_xor(psum, off, 32);
      l8[rr] = l8[rr] * corr + psum;
      m8[rr] = mnew;
#pragma unroll
      for (int t = 0; t < 4; ++t) oacc[t][rr] *= corr;
    }

    // ---- O += P @ V : P via per-wave LDS round-trip, V^T contiguous ----
    v16bf pfrag[2];
#pragma unroll
    for (int c = 0; c < 2; ++c)
      pfrag[c] = load_afrag(&lp[wave][lane & 15][c * 32 + baseA]);

    v16bf vf[8];
#pragma unroll
    for (int t = 0; t < 4; ++t) {
      const int n = t * 16 + (lane & 15);  // hd index = row of lvT
#pragma unroll
      for (int c = 0; c < 2; ++c) {
        const int kvoff = c * 32 + ((lane < 16) ? 0 : 16);
        vf[t * 2 + c] = *(const v16bf*)&lvT[n][kvoff];
      }
    }
#pragma unroll
    for (int t = 0; t < 4; ++t)
#pragma unroll
      for (int c = 0; c < 2; ++c)
        oacc[t] = wmma_bf16(pfrag[c], vf[t * 2 + c], oacc[t]);

    __syncthreads();  // before next tile overwrites lk/lvT
  }

  // ---- epilogue: normalize by l and write both output views ----
#pragma unroll
  for (int t = 0; t < 4; ++t) {
    const int hd = t * 16 + (lane & 15);
#pragma unroll
    for (int rr = 0; rr < 8; ++rr) {
      const int qrow = qb * 64 + wave * 16 + rr + ((lane >> 4) << 3);
      const float val = oacc[t][rr] / l8[rr];
      ctx[(((size_t)b * H_ + h) * S_ + qrow) * HD_ + hd] = val;
      attn[((size_t)b * S_ + qrow) * D_ + h * HD_ + hd] = val;
    }
  }
}

// ---------------------------------------------------------------------------
extern "C" void kernel_launch(void* const* d_in, const int* in_sizes, int n_in,
                              void* d_out, int out_size, void* d_ws,
                              size_t ws_size, hipStream_t stream) {
  const float* query = (const float*)d_in[0];
  const float* key   = (const float*)d_in[1];
  const float* value = (const float*)d_in[2];
  const float* Wq_w  = (const float*)d_in[3];
  const float* Wq_b  = (const float*)d_in[4];
  const float* Wk_w  = (const float*)d_in[5];
  const float* Wk_b  = (const float*)d_in[6];
  const float* Wv_w  = (const float*)d_in[7];
  const float* Wv_b  = (const float*)d_in[8];

  const size_t per = (size_t)B_ * H_ * S_ * HD_;  // 4M elems per matrix
  bf16* wsq = (bf16*)d_ws;                        // 8 MB
  bf16* wsk = wsq + per;                          // 8 MB
  bf16* wsv = wsk + per;                          // 8 MB (24 MB total)

  float* ctx  = (float*)d_out;   // context  [B,H,S,HD]
  float* attn = ctx + per;       // attn_out [B,S,D]

  dim3 gA(M_ / 64, D_ / 64, 3);
  qkv_proj_kernel<<<gA, 128, 0, stream>>>(query, key, value,
                                          Wq_w, Wq_b, Wk_w, Wk_b, Wv_w, Wv_b,
                                          wsq, wsk, wsv);

  dim3 gB(S_ / 64, H_, B_);
  flash_attn_kernel<<<gB, 128, 0, stream>>>(wsq, wsk, wsv, ctx, attn);
}
